// JpegCompression_82660940579061
// MI455X (gfx1250) — compile-verified, matrix operand order checked by reading, not run on hardware
//
#include <hip/hip_runtime.h>
#include <math.h>

// ---------------------------------------------------------------------------
// Fused JPEG-compression pipeline for MI455X (gfx1250, wave32).
//
//   rgb2yuv -> blockwise 8x8 DCT -> zigzag mask -> 8x8 IDCT -> yuv2rgb
//
// One wave handles NP=4 adjacent 16x16 pixel patches (each = four 8x8 tiles)
// across all 3 channels, entirely in registers:
//   D16 = diag(D, D)   (16x16 block-diagonal DCT operator)
//   Y   = D16 * X * D16^T            (two v_wmma_f32_16x16x32_f16, K padded)
//   Y  *= mask                       (in-lane, mask8[v][lane&7])
//   W   = Z^T = ID16 * Y^T * ID16^T  (two more WMMAs; C-layout of Y ==
//                                     A-layout of Y^T -> no transpose needed)
// C-layout accumulators convert in-lane to the next GEMM's B-operand, and
// diag(D,D) has identical register images as A-operand and as B-operand of
// its transpose -> zero LDS, zero shuffles. Constants (DCT rows via __cosf
// TRANS ops, zigzag masks via closed-form rank) are built once per wave and
// amortized over the 4-patch loop. Memory-bound: ~100 MB total traffic,
// ~4.3 us floor at 23.3 TB/s; per-patch cost is 12 WMMAs + 12 b128 mem ops.
// ---------------------------------------------------------------------------

typedef __attribute__((ext_vector_type(16))) _Float16 v16h;
typedef __attribute__((ext_vector_type(8)))  float    v8f;
typedef __attribute__((ext_vector_type(4)))  float    v4f;

#define WMMA16(A, B, C) \
  __builtin_amdgcn_wmma_f32_16x16x32_f16(false, (A), false, (B), (short)0, (C), false, false)

#define NPATCH_PER_WAVE 4

// Convert a C-layout f32 accumulator into the next GEMM's f16 operand
// (elements 0..7 live, 8..15 = zero-padding of K 16..31).
__device__ __forceinline__ v16h cvt_acc(v8f a) {
  v16h r;
#pragma unroll
  for (int j = 0; j < 8; ++j) {
    r[j]     = (_Float16)a[j];
    r[j + 8] = (_Float16)0.0f;
  }
  return r;
}

__global__ __launch_bounds__(256) void jpeg_fused_wmma(const float* __restrict__ in,
                                                       float* __restrict__ out,
                                                       int nWave) {
  const int lane = threadIdx.x & 31;
  const int wv   = threadIdx.x >> 5;
  const int gw   = blockIdx.x * 8 + wv;           // global wave id, uniform per wave
  if (gw >= nWave) return;                        // whole-wave exit: EXEC all-1s for WMMA

  const int l16 = lane & 15;
  const int hi  = lane >> 4;
  const int r8  = lane & 7;

  // ==== once-per-wave constants ===========================================
  // dA : diag(D,D)  in A-layout == diag(D,D)^T in B-layout (identical image)
  // idA: diag(ID,ID) likewise.  f16 targets -> __cosf (v_cos_f32) is enough.
  const bool diagOn = (((lane >> 3) & 1) == ((lane >> 4) & 1));
  const float PI8 = 0.39269908169872414f;         // pi/8
  v16h dA, idA;
#pragma unroll
  for (int j = 0; j < 8; ++j) {
    float dv = 0.0f, iv = 0.0f;
    if (diagOn) {
      dv = __cosf(PI8 * (j + 0.5f) * (float)r8);                                   // D[r8][j]
      iv = (__cosf(PI8 * ((float)r8 + 0.5f) * j) - (j == 0 ? 0.5f : 0.0f)) * 0.25f; // ID[r8][j]
    }
    dA[j]      = (_Float16)dv;
    idA[j]     = (_Float16)iv;
    dA[j + 8]  = (_Float16)0.0f;
    idA[j + 8] = (_Float16)0.0f;
  }

  // Zigzag masks: mask8[v][r8] via closed-form zigzag rank.
  float msk[3][8];
#pragma unroll
  for (int v = 0; v < 8; ++v) {
    const int s    = v + r8;
    const int rank = (s * (s + 1)) / 2 + ((s & 1) ? (s - r8) : r8);
    msk[0][v] = (rank < 25) ? 1.0f : 0.0f;   // Y keeps 25
    msk[1][v] = (rank < 9)  ? 1.0f : 0.0f;   // U keeps 9
    msk[2][v] = msk[1][v];                   // V keeps 9
  }

  const float R2Y[3][3] = {{ 0.299f,    0.587f,    0.114f   },
                           {-0.14713f, -0.28886f,  0.436f   },
                           { 0.615f,   -0.51499f, -0.10001f }};
  const float Y2R[3][3] = {{ 1.0f,  0.0f,      1.13983f },
                           { 1.0f, -0.39465f, -0.5806f  },
                           { 1.0f,  2.03211f,  0.0f     }};

  // ==== 4 adjacent patches: a 16x64 strip (same image row of patches) =====
  const size_t HW    = 512u * 512u;
  const int patch0   = gw * NPATCH_PER_WAVE;      // px0 = patch0 & 31 (multiple of 4)
  const int b        = patch0 >> 10;              // 1024 patches per image (32x32)
  const int py       = (patch0 >> 5) & 31;
  const int px0      = patch0 & 31;
  const int row      = py * 16 + l16;
  // lane covers (row, col0 + 16*i .. +7) for patch i
  const size_t base0 = (size_t)b * 3 * HW + (size_t)row * 512
                     + (size_t)(px0 * 16 + hi * 8);

#pragma unroll
  for (int i = 0; i < NPATCH_PER_WAVE; ++i) {
    const size_t base = base0 + (size_t)(16 * i);

    // ---- load: lane holds row l16, cols 8*hi..8*hi+7 of patch i ----------
    float rr[8], gg[8], bb[8];
    {
      const v4f* pR = reinterpret_cast<const v4f*>(in + base);
      const v4f* pG = reinterpret_cast<const v4f*>(in + base + HW);
      const v4f* pB = reinterpret_cast<const v4f*>(in + base + 2 * HW);
      v4f R0 = pR[0], R1 = pR[1];
      v4f G0 = pG[0], G1 = pG[1];
      v4f B0 = pB[0], B1 = pB[1];
#pragma unroll
      for (int j = 0; j < 4; ++j) {
        rr[j] = R0[j]; rr[j + 4] = R1[j];
        gg[j] = G0[j]; gg[j + 4] = G1[j];
        bb[j] = B0[j]; bb[j + 4] = B1[j];
      }
    }

    // ---- per-channel 4-GEMM pipeline --------------------------------------
    v8f W[3];
#pragma unroll
    for (int c = 0; c < 3; ++c) {
      // rgb2yuv pointwise, straight into the A-operand (X in A-layout)
      v16h aX;
#pragma unroll
      for (int j = 0; j < 8; ++j) {
        float y = R2Y[c][0] * rr[j] + R2Y[c][1] * gg[j] + R2Y[c][2] * bb[j];
        aX[j]     = (_Float16)y;
        aX[j + 8] = (_Float16)0.0f;
      }
      const v8f z = {};
      // P = X * D16^T
      v8f  P  = WMMA16(aX, dA, z);
      // Y = D16 * P     (C-layout of P == B-layout of P, in-lane convert)
      v16h bP = cvt_acc(P);
      v8f  Yc = WMMA16(dA, bP, z);
      // mask + feed Y^T as A-operand (C-layout of Y == A-layout of Y^T)
      v16h aY;
#pragma unroll
      for (int v = 0; v < 8; ++v) {
        aY[v]     = (_Float16)(Yc[v] * msk[c][v]);
        aY[v + 8] = (_Float16)0.0f;
      }
      // Q = Y^T * ID16^T
      v8f  Q  = WMMA16(aY, idA, z);
      // W = ID16 * Q = Z^T  -> lane holds row l16 of Z, cols 8*hi+v
      v16h bQ = cvt_acc(Q);
      W[c] = WMMA16(idA, bQ, z);
    }

    // ---- yuv2rgb pointwise + streamed stores (same coalesced pattern) -----
#pragma unroll
    for (int c = 0; c < 3; ++c) {
      v4f o0, o1;
#pragma unroll
      for (int v = 0; v < 4; ++v) {
        o0[v] = Y2R[c][0] * W[0][v]     + Y2R[c][1] * W[1][v]     + Y2R[c][2] * W[2][v];
        o1[v] = Y2R[c][0] * W[0][v + 4] + Y2R[c][1] * W[1][v + 4] + Y2R[c][2] * W[2][v + 4];
      }
      v4f* po = reinterpret_cast<v4f*>(out + base + (size_t)c * HW);
      __builtin_nontemporal_store(o0, po);
      __builtin_nontemporal_store(o1, po + 1);
    }
  }
}

extern "C" void kernel_launch(void* const* d_in, const int* in_sizes, int n_in,
                              void* d_out, int out_size, void* d_ws, size_t ws_size,
                              hipStream_t stream) {
  (void)in_sizes; (void)n_in; (void)out_size; (void)d_ws; (void)ws_size;
  const float* steg = (const float*)d_in[0];     // [16, 3, 512, 512] f32
  float*       outp = (float*)d_out;             // [16, 3, 512, 512] f32

  const int nPatch = 16 * 32 * 32;               // 16384 16x16 patches
  const int nWave  = nPatch / NPATCH_PER_WAVE;   // 4096 waves
  dim3 block(256);                               // 8 waves per workgroup
  dim3 grid(nWave / 8);                          // 512 workgroups
  jpeg_fused_wmma<<<grid, block, 0, stream>>>(steg, outp, nWave);
}
